// MultiheadCustomAttention_6107443495566
// MI455X (gfx1250) — matrix operand order repaired and checked by reading, not compiled, not run on hardware
//
#include <hip/hip_runtime.h>
#include <hip/hip_bf16.h>

// ---------------------------------------------------------------------------
// Problem constants: L=S=1024, B=4, E=1024, H=16, HD=64, BH=64, M=256.
// ---------------------------------------------------------------------------
#define CL   1024
#define CS   1024
#define CB   4
#define CE   1024
#define CH   16
#define CHD  64
#define CBH  64
#define CM   256
#define CROWS 4096

typedef __attribute__((ext_vector_type(16))) _Float16 v16h;
typedef __attribute__((ext_vector_type(8)))  float    v8f;
typedef __attribute__((ext_vector_type(4)))  int      v4i;

// ---------------------------------------------------------------------------
// Async global->LDS copy (CDNA5 GLOBAL_LOAD_ASYNC_TO_LDS_B128, ASYNCcnt) with
// a vectorized synchronous fallback if the builtin is absent on this toolchain.
// Probe-confirmed signature: (int4 __global*, int4 __local*, imm offset, imm cpol)
// ---------------------------------------------------------------------------
#if defined(__has_builtin)
#if __has_builtin(__builtin_amdgcn_global_load_async_to_lds_b128) && \
    __has_builtin(__builtin_amdgcn_s_wait_asynccnt)
#define USE_ASYNC 1
#endif
#endif

__device__ inline void cp16B(_Float16* l, const _Float16* g) {
#ifdef USE_ASYNC
  __builtin_amdgcn_global_load_async_to_lds_b128(
      (__attribute__((address_space(1))) v4i*)(void*)const_cast<_Float16*>(g),
      (__attribute__((address_space(3))) v4i*)(void*)l, 0, 0);
#else
  *(uint4*)l = *(const uint4*)g;
#endif
}

__device__ inline void cp_wait_async() {
#ifdef USE_ASYNC
  __builtin_amdgcn_s_wait_asynccnt(0);
#endif
}

// pack 4 f32 -> 4 f16 and store as one 8-byte LDS write
__device__ inline void st4h(_Float16* dst, float4 f) {
  union { _Float16 h[4]; unsigned long long u; } x;
  x.h[0] = (_Float16)f.x; x.h[1] = (_Float16)f.y;
  x.h[2] = (_Float16)f.z; x.h[3] = (_Float16)f.w;
  *(unsigned long long*)dst = x.u;
}

// pack 8 f32 -> 8 f16, one 16-byte store
__device__ inline void st8h(_Float16* dst, float4 a, float4 b) {
  union { _Float16 h[8]; uint4 u; } x;
  x.h[0] = (_Float16)a.x; x.h[1] = (_Float16)a.y;
  x.h[2] = (_Float16)a.z; x.h[3] = (_Float16)a.w;
  x.h[4] = (_Float16)b.x; x.h[5] = (_Float16)b.y;
  x.h[6] = (_Float16)b.z; x.h[7] = (_Float16)b.w;
  *(uint4*)dst = x.u;
}

// ---------------------------------------------------------------------------
// WMMA fragment helpers (gfx1250 wave32, v_wmma_f32_16x16x32_f16)
// ---------------------------------------------------------------------------
__device__ inline v16h frag_a(const _Float16* base, int ldh, int row, int kbase) {
  const int lane = threadIdx.x & 31;
  const int hi = (lane & 16) ? 16 : 0;
  const unsigned short* p = (const unsigned short*)(base + row * ldh + kbase);
  union { v16h v; unsigned int u[8]; } f;
#pragma unroll
  for (int g = 0; g < 4; ++g) f.u[g]     = *(const unsigned int*)(p + hi + 2 * g);
#pragma unroll
  for (int g = 0; g < 4; ++g) f.u[4 + g] = *(const unsigned int*)(p + hi + 8 + 2 * g);
  return f.v;
}

__device__ inline v16h frag_b_nk(const _Float16* base, int ldh, int nbase, int kbase) {
  const int lane = threadIdx.x & 31;
  const unsigned short* p = (const unsigned short*)
      (base + (nbase + (lane & 15)) * ldh + kbase + ((lane & 16) ? 16 : 0));
  union { v16h v; unsigned int u[8]; } f;
#pragma unroll
  for (int g = 0; g < 8; ++g) f.u[g] = *(const unsigned int*)(p + 2 * g);
  return f.v;
}

__device__ inline v16h frag_b_kn(const _Float16* base, int ldh, int kbase, int nbase) {
  const int lane = threadIdx.x & 31;
  const unsigned short* p = (const unsigned short*)base;
  const int n  = nbase + (lane & 15);
  const int k0 = kbase + ((lane & 16) ? 16 : 0);
  union { v16h v; unsigned int u[8]; } f;
#pragma unroll
  for (int g = 0; g < 8; ++g) {
    unsigned int lo = p[(k0 + 2 * g) * ldh + n];
    unsigned int hi = p[(k0 + 2 * g + 1) * ldh + n];
    f.u[g] = lo | (hi << 16);
  }
  return f.v;
}

__device__ inline v8f wmma16(v16h a, v16h b, v8f c) {
  return __builtin_amdgcn_wmma_f32_16x16x32_f16(false, a, false, b, (short)0, c,
                                                false, false);
}

__device__ inline v8f vzero8() { v8f z = {0.f,0.f,0.f,0.f,0.f,0.f,0.f,0.f}; return z; }

// ---------------------------------------------------------------------------
// Kernel 1: fused q/k/v projection + bias (+scale, +interleaved rotary)
// ---------------------------------------------------------------------------
__global__ void __launch_bounds__(256)
k_proj_qkv(const float* __restrict__ qin, const float* __restrict__ kin,
           const float* __restrict__ vin, const float* __restrict__ w,
           const float* __restrict__ bias, const float* __restrict__ q_pe,
           const float* __restrict__ kv_pe,
           _Float16* __restrict__ qh, _Float16* __restrict__ kh,
           _Float16* __restrict__ vh) {
  const int which = blockIdx.z;
  const int row0 = blockIdx.y * 128;
  const int col0 = blockIdx.x * 64;
  const float* A = (which == 0) ? qin : (which == 1) ? kin : vin;

  __shared__ _Float16 As[128 * 32];
  __shared__ _Float16 Bs[64 * 32];
  const int t = threadIdx.x, wv = t >> 5, lane = t & 31;

  v8f acc[4];
#pragma unroll
  for (int i = 0; i < 4; ++i) acc[i] = vzero8();

  for (int kk = 0; kk < CE; kk += 32) {
    __syncthreads();
    // batch all global loads first (float4), then convert+store: keeps
    // several global_load_b128 in flight instead of load/wait per element.
    float4 av[4], bv[2];
#pragma unroll
    for (int i = 0; i < 4; ++i) {              // 128x32 A tile
      int v4 = i * 256 + t;
      int r = v4 >> 3, c = (v4 & 7) * 4;
      av[i] = *(const float4*)&A[(size_t)(row0 + r) * CE + kk + c];
    }
#pragma unroll
    for (int i = 0; i < 2; ++i) {              // 64x32 W tile [eo][ei]
      int v4 = i * 256 + t;
      int n = v4 >> 3, c = (v4 & 7) * 4;
      bv[i] = *(const float4*)&w[(size_t)(which * CE + col0 + n) * CE + kk + c];
    }
#pragma unroll
    for (int i = 0; i < 4; ++i) st4h(&As[(i * 256 + t) * 4], av[i]);
#pragma unroll
    for (int i = 0; i < 2; ++i) st4h(&Bs[(i * 256 + t) * 4], bv[i]);
    __syncthreads();

    v16h a = frag_a(As, 32, wv * 16 + (lane & 15), 0);
#pragma unroll
    for (int tt = 0; tt < 4; ++tt) {
      v16h b = frag_b_nk(Bs, 32, tt * 16, 0);
      acc[tt] = wmma16(a, b, acc[tt]);
    }
  }

  const float scale = (which == 0) ? 0.125f : 1.0f;
  const float* pe = (which == 0) ? q_pe : kv_pe;
  _Float16* dst = (which == 0) ? qh : (which == 1) ? kh : vh;

#pragma unroll
  for (int tt = 0; tt < 4; ++tt) {
#pragma unroll
    for (int r = 0; r < 8; ++r) {
      int row = row0 + wv * 16 + r + ((lane & 16) ? 8 : 0);  // token row l*B+b
      int e = col0 + tt * 16 + (lane & 15);
      int l = row >> 2, b = row & 3;
      float y = (acc[tt][r] + bias[which * CE + e]) * scale;
      float out;
      if (which < 2) {
        float yp = __shfl_xor(y, 1, 32);       // neighbor column e^1
        float x2 = (e & 1) ? yp : -yp;         // even: -x[e+1], odd: x[e-1]
        size_t pidx = (((size_t)b * CL + l) * CE + e) * 2;
        out = y * pe[pidx] + x2 * pe[pidx + 1];
      } else {
        out = y;
      }
      int h = e >> 6, hd = e & 63;
      dst[(((size_t)(b * CH + h)) * CL + l) * CHD + hd] = (_Float16)out;
    }
  }
}

// ---------------------------------------------------------------------------
// Kernel 2: reshape k_mem/v_mem (M,B,E) f32 -> head-major [bh][m][hd] f16
// ---------------------------------------------------------------------------
__global__ void __launch_bounds__(256)
k_mem_reshape(const float* __restrict__ km, const float* __restrict__ vm,
              _Float16* __restrict__ kmh, _Float16* __restrict__ vmh) {
  int v8 = blockIdx.x * 256 + threadIdx.x;           // over (M*B*E)/8 = 131072
  size_t idx = (size_t)v8 * 8;
  int e = (int)(idx & (CE - 1)), b = (int)((idx >> 10) & 3), m = (int)(idx >> 12);
  int h = e >> 6, hd = e & 63;
  size_t d = (((size_t)(b * CH + h)) * CM + m) * CHD + hd;
  float4 a0 = *(const float4*)&km[idx];
  float4 a1 = *(const float4*)&km[idx + 4];
  float4 b0 = *(const float4*)&vm[idx];
  float4 b1 = *(const float4*)&vm[idx + 4];
  st8h(&kmh[d], a0, a1);
  st8h(&vmh[d], b0, b1);
}

// ---------------------------------------------------------------------------
// Kernel 3: column-softmax stats (online max/sum over L per key column)
// ---------------------------------------------------------------------------
__global__ void __launch_bounds__(256)
k_colstats(const _Float16* __restrict__ qh, const _Float16* __restrict__ kh,
           float* __restrict__ colmax, float* __restrict__ colsum) {
  const int bh = blockIdx.y, s0 = blockIdx.x * 64;
  const int t = threadIdx.x, wv = t >> 5, lane = t & 31;

  __shared__ _Float16 Ks[64 * 64];
  __shared__ _Float16 Qs[128 * 64];
  __shared__ float cm[64], cs[64], nm[64];
  __shared__ float wmax[8][64], wsum[8][64];

#pragma unroll
  for (int i = 0; i < 2; ++i) {                      // 64x64 K tile (async)
    int v8 = i * 256 + t;
    int r = v8 >> 3, c = (v8 & 7) * 8;
    cp16B(&Ks[r * 64 + c], &kh[((size_t)bh * CS + s0 + r) * CHD + c]);
  }
  if (t < 64) { cm[t] = -1e30f; cs[t] = 0.f; }
  cp_wait_async();
  __syncthreads();

  for (int l0 = 0; l0 < CL; l0 += 128) {
    __syncthreads();
#pragma unroll
    for (int i = 0; i < 4; ++i) {                    // 128x64 Q tile (async)
      int v8 = i * 256 + t;
      int r = v8 >> 3, c = (v8 & 7) * 8;
      cp16B(&Qs[r * 64 + c], &qh[((size_t)bh * CL + l0 + r) * CHD + c]);
    }
    cp_wait_async();
    __syncthreads();

    v16h a0 = frag_a(Qs, 64, wv * 16 + (lane & 15), 0);
    v16h a1 = frag_a(Qs, 64, wv * 16 + (lane & 15), 32);
    v8f sc[4];
#pragma unroll
    for (int tt = 0; tt < 4; ++tt) {
      v16h b0 = frag_b_nk(Ks, 64, tt * 16, 0);
      v16h b1 = frag_b_nk(Ks, 64, tt * 16, 32);
      sc[tt] = wmma16(a0, b0, vzero8());
      sc[tt] = wmma16(a1, b1, sc[tt]);
    }
#pragma unroll
    for (int tt = 0; tt < 4; ++tt) {
      float pm = sc[tt][0];
#pragma unroll
      for (int r = 1; r < 8; ++r) pm = fmaxf(pm, sc[tt][r]);
      pm = fmaxf(pm, __shfl_xor(pm, 16, 32));
      if (lane < 16) wmax[wv][tt * 16 + lane] = pm;
    }
    __syncthreads();
    if (t < 64) {
      float m = wmax[0][t];
#pragma unroll
      for (int w8 = 1; w8 < 8; ++w8) m = fmaxf(m, wmax[w8][t]);
      float newm = fmaxf(cm[t], m);
      cs[t] *= __expf(cm[t] - newm);
      cm[t] = newm; nm[t] = newm;
    }
    __syncthreads();
#pragma unroll
    for (int tt = 0; tt < 4; ++tt) {
      float ps = 0.f;
#pragma unroll
      for (int r = 0; r < 8; ++r) ps += __expf(sc[tt][r] - nm[tt * 16 + (lane & 15)]);
      ps += __shfl_xor(ps, 16, 32);
      if (lane < 16) wsum[wv][tt * 16 + lane] = ps;
    }
    __syncthreads();
    if (t < 64) {
      float s = 0.f;
#pragma unroll
      for (int w8 = 0; w8 < 8; ++w8) s += wsum[w8][t];
      cs[t] += s;
    }
  }
  __syncthreads();
  if (t < 64) {
    colmax[(size_t)bh * CS + s0 + t] = cm[t];
    colsum[(size_t)bh * CS + s0 + t] = cs[t];
  }
}

// ---------------------------------------------------------------------------
// Kernel 4: main attention: w = exp(sc-cmax)/csum + 1e-8; out = (W@V)/rowsum(W)
// ---------------------------------------------------------------------------
__global__ void __launch_bounds__(256)
k_attn_main(const _Float16* __restrict__ qh, const _Float16* __restrict__ kh,
            const _Float16* __restrict__ vh, const float* __restrict__ colmax,
            const float* __restrict__ colsum, float* __restrict__ attn_main) {
  const int bh = blockIdx.y, l0 = blockIdx.x * 128;
  const int t = threadIdx.x, wv = t >> 5, lane = t & 31;

  __shared__ _Float16 Qs[128 * 64];
  __shared__ _Float16 Kc[64 * 64];
  __shared__ _Float16 Vc[64 * 64];
  __shared__ _Float16 Ps[128 * 64];
  __shared__ float cmx[64], csm[64], den[128];

#pragma unroll
  for (int i = 0; i < 4; ++i) {
    int v8 = i * 256 + t;
    int r = v8 >> 3, c = (v8 & 7) * 8;
    cp16B(&Qs[r * 64 + c], &qh[((size_t)bh * CL + l0 + r) * CHD + c]);
  }
  if (t < 128) den[t] = 0.f;

  v8f acc[4];
#pragma unroll
  for (int i = 0; i < 4; ++i) acc[i] = vzero8();

  for (int s0 = 0; s0 < CS; s0 += 64) {
    __syncthreads();
#pragma unroll
    for (int i = 0; i < 2; ++i) {
      int v8 = i * 256 + t;
      int r = v8 >> 3, c = (v8 & 7) * 8;
      cp16B(&Kc[r * 64 + c], &kh[((size_t)bh * CS + s0 + r) * CHD + c]);
      cp16B(&Vc[r * 64 + c], &vh[((size_t)bh * CS + s0 + r) * CHD + c]);
    }
    if (t < 64) {
      cmx[t] = colmax[(size_t)bh * CS + s0 + t];
      csm[t] = colsum[(size_t)bh * CS + s0 + t];
    }
    cp_wait_async();
    __syncthreads();

    v16h a0 = frag_a(Qs, 64, wv * 16 + (lane & 15), 0);
    v16h a1 = frag_a(Qs, 64, wv * 16 + (lane & 15), 32);
#pragma unroll
    for (int tt = 0; tt < 4; ++tt) {
      v16h b0 = frag_b_nk(Kc, 64, tt * 16, 0);
      v16h b1 = frag_b_nk(Kc, 64, tt * 16, 32);
      v8f sc = wmma16(a0, b0, vzero8());
      sc = wmma16(a1, b1, sc);
      int col = tt * 16 + (lane & 15);
#pragma unroll
      for (int r = 0; r < 8; ++r) {
        int row = wv * 16 + r + ((lane & 16) ? 8 : 0);
        float p = __expf(sc[r] - cmx[col]) / csm[col] + 1e-8f;
        Ps[row * 64 + col] = (_Float16)p;
      }
    }
    __syncthreads();
    if (t < 128) {                                   // row denominator
      float s = 0.f;
#pragma unroll
      for (int j = 0; j < 64; ++j) s += (float)Ps[t * 64 + j];
      den[t] += s;
    }
    v16h pa0 = frag_a(Ps, 64, wv * 16 + (lane & 15), 0);
    v16h pa1 = frag_a(Ps, 64, wv * 16 + (lane & 15), 32);
#pragma unroll
    for (int tt = 0; tt < 4; ++tt) {
      v16h b0 = frag_b_kn(Vc, 64, 0, tt * 16);
      v16h b1 = frag_b_kn(Vc, 64, 32, tt * 16);
      acc[tt] = wmma16(pa0, b0, acc[tt]);
      acc[tt] = wmma16(pa1, b1, acc[tt]);
    }
  }
  __syncthreads();
#pragma unroll
  for (int tt = 0; tt < 4; ++tt) {
#pragma unroll
    for (int r = 0; r < 8; ++r) {
      int rl = wv * 16 + r + ((lane & 16) ? 8 : 0);
      int hd = tt * 16 + (lane & 15);
      attn_main[((size_t)bh * CL + l0 + rl) * CHD + hd] = acc[tt][r] / den[rl];
    }
  }
}

// ---------------------------------------------------------------------------
// Kernel 5: memory attention (flash over M=256) + sigmoid gate + combine
// ---------------------------------------------------------------------------
__global__ void __launch_bounds__(256)
k_attn_mem_gate(const _Float16* __restrict__ qh, const _Float16* __restrict__ kmh,
                const _Float16* __restrict__ vmh, const float* __restrict__ gate_attn,
                const float* __restrict__ attn_main, _Float16* __restrict__ attn_comb) {
  const int bh = blockIdx.y, l0 = blockIdx.x * 128;
  const int t = threadIdx.x, wv = t >> 5, lane = t & 31;

  __shared__ _Float16 Qs[128 * 64];
  __shared__ _Float16 Kc[64 * 64];
  __shared__ _Float16 Vc[64 * 64];
  __shared__ _Float16 Ps[128 * 64];

#pragma unroll
  for (int i = 0; i < 4; ++i) {
    int v8 = i * 256 + t;
    int r = v8 >> 3, c = (v8 & 7) * 8;
    cp16B(&Qs[r * 64 + c], &qh[((size_t)bh * CL + l0 + r) * CHD + c]);
  }

  float rmax[8], rsum[8];
#pragma unroll
  for (int r = 0; r < 8; ++r) { rmax[r] = -1e30f; rsum[r] = 0.f; }
  v8f oacc[4];
#pragma unroll
  for (int i = 0; i < 4; ++i) oacc[i] = vzero8();

  for (int m0 = 0; m0 < CM; m0 += 64) {
    __syncthreads();
#pragma unroll
    for (int i = 0; i < 2; ++i) {
      int v8 = i * 256 + t;
      int r = v8 >> 3, c = (v8 & 7) * 8;
      cp16B(&Kc[r * 64 + c], &kmh[((size_t)bh * CM + m0 + r) * CHD + c]);
      cp16B(&Vc[r * 64 + c], &vmh[((size_t)bh * CM + m0 + r) * CHD + c]);
    }
    cp_wait_async();
    __syncthreads();

    v16h a0 = frag_a(Qs, 64, wv * 16 + (lane & 15), 0);
    v16h a1 = frag_a(Qs, 64, wv * 16 + (lane & 15), 32);
    v8f sc[4];
#pragma unroll
    for (int tt = 0; tt < 4; ++tt) {
      v16h b0 = frag_b_nk(Kc, 64, tt * 16, 0);
      v16h b1 = frag_b_nk(Kc, 64, tt * 16, 32);
      sc[tt] = wmma16(a0, b0, vzero8());
      sc[tt] = wmma16(a1, b1, sc[tt]);
    }
#pragma unroll
    for (int r = 0; r < 8; ++r) {
      float tm = fmaxf(fmaxf(sc[0][r], sc[1][r]), fmaxf(sc[2][r], sc[3][r]));
      tm = fmaxf(tm, __shfl_xor(tm, 1, 32));
      tm = fmaxf(tm, __shfl_xor(tm, 2, 32));
      tm = fmaxf(tm, __shfl_xor(tm, 4, 32));
      tm = fmaxf(tm, __shfl_xor(tm, 8, 32));
      float nmx = fmaxf(rmax[r], tm);
      float scl = __expf(rmax[r] - nmx);
      rsum[r] *= scl;
      rmax[r] = nmx;
#pragma unroll
      for (int tt = 0; tt < 4; ++tt) oacc[tt][r] *= scl;
      float ps = 0.f;
#pragma unroll
      for (int tt = 0; tt < 4; ++tt) {
        float p = __expf(sc[tt][r] - nmx);
        sc[tt][r] = p;
        ps += p;
      }
      ps += __shfl_xor(ps, 1, 32);
      ps += __shfl_xor(ps, 2, 32);
      ps += __shfl_xor(ps, 4, 32);
      ps += __shfl_xor(ps, 8, 32);
      rsum[r] += ps;
    }
#pragma unroll
    for (int tt = 0; tt < 4; ++tt) {
      int col = tt * 16 + (lane & 15);
#pragma unroll
      for (int r = 0; r < 8; ++r) {
        int row = wv * 16 + r + ((lane & 16) ? 8 : 0);
        Ps[row * 64 + col] = (_Float16)sc[tt][r];
      }
    }
    __syncthreads();
    v16h pa0 = frag_a(Ps, 64, wv * 16 + (lane & 15), 0);
    v16h pa1 = frag_a(Ps, 64, wv * 16 + (lane & 15), 32);
#pragma unroll
    for (int tt = 0; tt < 4; ++tt) {
      v16h b0 = frag_b_kn(Vc, 64, 0, tt * 16);
      v16h b1 = frag_b_kn(Vc, 64, 32, tt * 16);
      oacc[tt] = wmma16(pa0, b0, oacc[tt]);
      oacc[tt] = wmma16(pa1, b1, oacc[tt]);
    }
  }

  const int b = bh >> 4, h = bh & 15;
  const float g = 1.f / (1.f + __expf(-gate_attn[h]));
#pragma unroll
  for (int tt = 0; tt < 4; ++tt) {
#pragma unroll
    for (int r = 0; r < 8; ++r) {
      int rl = wv * 16 + r + ((lane & 16) ? 8 : 0);
      int l = l0 + rl;
      int hd = tt * 16 + (lane & 15);
      float amem = oacc[tt][r] / rsum[r];
      float amain = attn_main[((size_t)bh * CL + l) * CHD + hd];
      float val = g * amem + (1.f - g) * amain;
      attn_comb[((size_t)l * CB + b) * CE + h * CHD + hd] = (_Float16)val;
    }
  }
}

// ---------------------------------------------------------------------------
// Kernel 6: output projection (4096x1024 f16) @ W^T + bias -> f32
// ---------------------------------------------------------------------------
__global__ void __launch_bounds__(256)
k_out_proj(const _Float16* __restrict__ ac, const float* __restrict__ w,
           const float* __restrict__ bias, float* __restrict__ out) {
  const int row0 = blockIdx.y * 128;
  const int col0 = blockIdx.x * 64;
  __shared__ _Float16 As[128 * 32];
  __shared__ _Float16 Bs[64 * 32];
  const int t = threadIdx.x, wv = t >> 5, lane = t & 31;

  v8f acc[4];
#pragma unroll
  for (int i = 0; i < 4; ++i) acc[i] = vzero8();

  for (int kk = 0; kk < CE; kk += 32) {
    __syncthreads();
#pragma unroll
    for (int i = 0; i < 2; ++i) {              // A tile f16: async b128 copy
      int v8 = i * 256 + t;
      int r = v8 >> 2, c = (v8 & 3) * 8;
      cp16B(&As[r * 32 + c], &ac[(size_t)(row0 + r) * CE + kk + c]);
    }
    float4 bv[2];
#pragma unroll
    for (int i = 0; i < 2; ++i) {              // W tile f32 -> f16
      int v4 = i * 256 + t;
      int n = v4 >> 3, c = (v4 & 7) * 4;
      bv[i] = *(const float4*)&w[(size_t)(col0 + n) * CE + kk + c];
    }
#pragma unroll
    for (int i = 0; i < 2; ++i) st4h(&Bs[(i * 256 + t) * 4], bv[i]);
    cp_wait_async();
    __syncthreads();

    v16h a = frag_a(As, 32, wv * 16 + (lane & 15), 0);
#pragma unroll
    for (int tt = 0; tt < 4; ++tt) {
      v16h b = frag_b_nk(Bs, 32, tt * 16, 0);
      acc[tt] = wmma16(a, b, acc[tt]);
    }
  }
#pragma unroll
  for (int tt = 0; tt < 4; ++tt) {
#pragma unroll
    for (int r = 0; r < 8; ++r) {
      int row = row0 + wv * 16 + r + ((lane & 16) ? 8 : 0);
      int e = col0 + tt * 16 + (lane & 15);
      out[(size_t)row * CE + e] = acc[tt][r] + bias[e];
    }
  }
}

// ---------------------------------------------------------------------------
extern "C" void kernel_launch(void* const* d_in, const int* in_sizes, int n_in,
                              void* d_out, int out_size, void* d_ws, size_t ws_size,
                              hipStream_t stream) {
  const float* query  = (const float*)d_in[0];
  const float* key    = (const float*)d_in[1];
  const float* value  = (const float*)d_in[2];
  const float* ipw    = (const float*)d_in[3];
  const float* ipb    = (const float*)d_in[4];
  const float* opw    = (const float*)d_in[5];
  const float* opb    = (const float*)d_in[6];
  const float* q_pe   = (const float*)d_in[7];
  const float* kv_pe  = (const float*)d_in[8];
  const float* k_mem  = (const float*)d_in[9];
  const float* v_mem  = (const float*)d_in[10];
  const float* gate   = (const float*)d_in[11];
  float* out = (float*)d_out;

  char* ws = (char*)d_ws;
  size_t o = 0;
  _Float16* qh  = (_Float16*)(ws + o); o += (size_t)CBH * CL * CHD * 2;
  _Float16* kh  = (_Float16*)(ws + o); o += (size_t)CBH * CS * CHD * 2;
  _Float16* vh  = (_Float16*)(ws + o); o += (size_t)CBH * CS * CHD * 2;
  _Float16* kmh = (_Float16*)(ws + o); o += (size_t)CBH * CM * CHD * 2;
  _Float16* vmh = (_Float16*)(ws + o); o += (size_t)CBH * CM * CHD * 2;
  float* colmax = (float*)(ws + o);    o += (size_t)CBH * CS * 4;
  float* colsum = (float*)(ws + o);    o += (size_t)CBH * CS * 4;
  float* amain  = (float*)(ws + o);    o += (size_t)CBH * CL * CHD * 4;
  _Float16* acomb = (_Float16*)(ws + o); o += (size_t)CROWS * CE * 2;

  dim3 blk(256);
  k_proj_qkv<<<dim3(CE / 64, CROWS / 128, 3), blk, 0, stream>>>(
      query, key, value, ipw, ipb, q_pe, kv_pe, qh, kh, vh);
  k_mem_reshape<<<dim3((CM * CB * CE) / (256 * 8)), blk, 0, stream>>>(k_mem, v_mem, kmh, vmh);
  k_colstats<<<dim3(CS / 64, CBH), blk, 0, stream>>>(qh, kh, colmax, colsum);
  k_attn_main<<<dim3(CL / 128, CBH), blk, 0, stream>>>(qh, kh, vh, colmax, colsum, amain);
  k_attn_mem_gate<<<dim3(CL / 128, CBH), blk, 0, stream>>>(qh, kmh, vmh, gate, amain, acomb);
  k_out_proj<<<dim3(CE / 64, CROWS / 128), blk, 0, stream>>>(acomb, opw, opb, out);
}
